// Attention_26946624815764
// MI455X (gfx1250) — matrix operand order repaired and checked by reading, not compiled, not run on hardware
//
#include <hip/hip_runtime.h>
#include <hip/hip_bf16.h>

// ---------------------------------------------------------------------------
// MI455X (gfx1250) attention: bf16 WMMA everywhere, f32 accumulate.
//   k0: gemm<float,0>  : Q = Xq@Wq+bq (scaled 1/sqrt(128)) -> bf16 [B,H,TQ,128]
//   k1: gemm<float,0>  : K = Xkv@Wk+bk                     -> bf16 [B,H,TKV,128]
//   k2: gemm<float,2>  : V^T = (Xkv@Wv+bv)^T               -> bf16 [B,H,128,TKV]
//   k3: flash-attention, K/V^T tiles streamed with async global->LDS copies
//       (double-buffered), online softmax, mask + wipe     -> bf16 [B,TQ,1024]
//   k4: gemm<bf16,1>   : out = AO@Wo+bo                    -> f32  [B,TQ,1024]
// ---------------------------------------------------------------------------

typedef __attribute__((ext_vector_type(16))) __bf16 v16bf;
typedef __attribute__((ext_vector_type(8)))  float  v8f;
typedef __attribute__((ext_vector_type(4)))  int    v4i;

union Frag16 { v16bf f; uint4 q[2]; };
static_assert(sizeof(Frag16) == 32, "frag size");

#define WMMA_BF16(a, b, c) \
  __builtin_amdgcn_wmma_f32_16x16x32_bf16(false, (a), false, (b), (short)0, (c), false, false)

#define NB  4      // batch
#define NH  8      // heads
#define NTQ 512
#define NTKV 4096
#define ND  1024
#define NHD 128

// ---- async global->LDS 16B copy (CDNA5 GLOBAL_LOAD_ASYNC_TO_LDS_B128) ------
// Builtin signature (from hipcc diagnostic): (v4i addrspace(1)*, v4i addrspace(3)*, imm, imm)
typedef __attribute__((address_space(1))) v4i gas_v4i;
typedef __attribute__((address_space(3))) v4i las_v4i;

__device__ __forceinline__ void cp16_async(const __bf16* g, __bf16* l) {
#if defined(__gfx1250__) && __has_builtin(__builtin_amdgcn_global_load_async_to_lds_b128)
  __builtin_amdgcn_global_load_async_to_lds_b128(
      (gas_v4i*)(g), (las_v4i*)(l), 0, 0);
#else
  *(uint4*)l = *(const uint4*)g;
#endif
}
__device__ __forceinline__ void wait_async_all() {
#if defined(__gfx1250__) && __has_builtin(__builtin_amdgcn_global_load_async_to_lds_b128)
# if __has_builtin(__builtin_amdgcn_s_wait_asynccnt)
  __builtin_amdgcn_s_wait_asynccnt(0);
# else
  asm volatile("s_wait_asynccnt 0" ::: "memory");
# endif
#endif
}

// ---------------------------------------------------------------------------
// Tiled GEMM: C[M,1024] = A[M,1024] * W[1024,1024] + bias
// block = 256 threads (8 waves), tile 128x128, K-step 32.
// wave (wm = w%4, wn = w/4) computes a 32x64 sub-tile = 2x4 grid of 16x16 WMMAs.
// MODE 0: bf16, head-split dst[b][h][t][d] (n = h*128+d), scaled.
// MODE 1: f32 dst[m*1024+n].
// MODE 2: bf16, transposed head-split dst[b][h][d][t], scaled.
// ---------------------------------------------------------------------------
template <typename TA, int MODE>
__global__ __launch_bounds__(256) void gemm_bf16_wmma(
    const TA* __restrict__ A, const float* __restrict__ W,
    const float* __restrict__ bias, void* __restrict__ dstv,
    int T, float scale)
{
  __shared__ alignas(16) __bf16 As[128 * 32];   // [m][k] row-major
  __shared__ alignas(16) __bf16 Bs[128 * 32];   // [n][k] (transposed while staging)

  const int tid  = threadIdx.x;
  const int wav  = tid >> 5;
  const int lane = tid & 31;
  const int hlf  = lane >> 4;
  const int l16  = lane & 15;
  const int wm   = wav & 3;
  const int wn   = wav >> 2;
  const int mbase = blockIdx.x * 128;
  const int nbase = blockIdx.y * 128;

  v8f acc[2][4] = {};

  for (int kb = 0; kb < 1024; kb += 32) {
    __syncthreads();
    // ---- stage A tile (128x32) into LDS as bf16 ----
    if constexpr (sizeof(TA) == 4) {
      #pragma unroll
      for (int j = 0; j < 4; ++j) {
        int lin = tid * 16 + j * 4;
        int r = lin >> 5, k = lin & 31;
        float4 v = *(const float4*)(&A[(size_t)(mbase + r) * 1024 + kb + k]);
        As[r * 32 + k + 0] = (__bf16)v.x;  As[r * 32 + k + 1] = (__bf16)v.y;
        As[r * 32 + k + 2] = (__bf16)v.z;  As[r * 32 + k + 3] = (__bf16)v.w;
      }
    } else {
      #pragma unroll
      for (int j = 0; j < 2; ++j) {
        int lin = tid * 16 + j * 8;
        int r = lin >> 5, k = lin & 31;
        cp16_async((const __bf16*)&A[(size_t)(mbase + r) * 1024 + kb + k],
                   &As[r * 32 + k]);
      }
    }
    // ---- stage B tile (32x128) transposed into LDS [n][k] ----
    #pragma unroll
    for (int j = 0; j < 4; ++j) {
      int lin = tid * 16 + j * 4;
      int kr = lin >> 7, nc = lin & 127;
      float4 v = *(const float4*)(&W[(size_t)(kb + kr) * 1024 + nbase + nc]);
      Bs[(nc + 0) * 32 + kr] = (__bf16)v.x;
      Bs[(nc + 1) * 32 + kr] = (__bf16)v.y;
      Bs[(nc + 2) * 32 + kr] = (__bf16)v.z;
      Bs[(nc + 3) * 32 + kr] = (__bf16)v.w;
    }
    if constexpr (sizeof(TA) == 2) wait_async_all();
    __syncthreads();

    // ---- A fragments: row = lane%16; k runs {8*hlf..+7} and {16+8*hlf..+7} ----
    Frag16 af[2];
    #pragma unroll
    for (int mt = 0; mt < 2; ++mt) {
      const __bf16* p = &As[(wm * 32 + mt * 16 + l16) * 32 + hlf * 8];
      af[mt].q[0] = *(const uint4*)p;
      af[mt].q[1] = *(const uint4*)(p + 16);
    }
    // ---- B fragments: col = lane%16; k runs {8*hlf..+7} and {16+8*hlf..+7} ----
    Frag16 bfr[4];
    #pragma unroll
    for (int nt = 0; nt < 4; ++nt) {
      const __bf16* p = &Bs[(wn * 64 + nt * 16 + l16) * 32 + hlf * 8];
      bfr[nt].q[0] = *(const uint4*)p;
      bfr[nt].q[1] = *(const uint4*)(p + 16);
    }
    #pragma unroll
    for (int mt = 0; mt < 2; ++mt)
      #pragma unroll
      for (int nt = 0; nt < 4; ++nt)
        acc[mt][nt] = WMMA_BF16(af[mt].f, bfr[nt].f, acc[mt][nt]);
  }

  // ---- epilogue: C layout lane=col, vgpr i + 8*hlf = row ----
  #pragma unroll
  for (int mt = 0; mt < 2; ++mt) {
    #pragma unroll
    for (int nt = 0; nt < 4; ++nt) {
      #pragma unroll
      for (int i = 0; i < 8; ++i) {
        int m = mbase + wm * 32 + mt * 16 + i + 8 * hlf;
        int n = nbase + wn * 64 + nt * 16 + l16;
        float v = acc[mt][nt][i] + bias[n];
        if constexpr (MODE == 0) {
          v *= scale;
          int b = m / T, t = m % T, h = n >> 7, d = n & 127;
          ((__bf16*)dstv)[((((size_t)b * NH + h) * T + t) << 7) + d] = (__bf16)v;
        } else if constexpr (MODE == 2) {
          v *= scale;
          int b = m / T, t = m % T, h = n >> 7, d = n & 127;
          ((__bf16*)dstv)[(((size_t)b * NH + h) * NHD + d) * T + t] = (__bf16)v;
        } else {
          ((float*)dstv)[(size_t)m * 1024 + n] = v;
        }
      }
    }
  }
}

// ---------------------------------------------------------------------------
// Flash attention. grid = B*H*(TQ/128) blocks of 256 threads.
// Each wave: 16 q-rows x full hd=128 (8 accumulator tiles).
// KV streamed in 64-column tiles via double-buffered async global->LDS copies.
// V arrives pre-transposed ([b,h,d,t]) so both K and V^T tiles are contiguous
// 16B chunks. Q pre-scaled by 1/sqrt(128) in the projection.
// ---------------------------------------------------------------------------
__global__ __launch_bounds__(256) void attn_fwd_wmma(
    const __bf16* __restrict__ Q, const __bf16* __restrict__ K,
    const __bf16* __restrict__ Vt, const unsigned char* __restrict__ mask,
    __bf16* __restrict__ AO)
{
  __shared__ alignas(16) __bf16 Ks[2][64 * 128];   // [kv][hd]
  __shared__ alignas(16) __bf16 Vs[2][128 * 64];   // [hd][kv]
  __shared__ alignas(16) __bf16 Ps[8][16 * 64];    // per-wave P staging

  const int tid  = threadIdx.x;
  const int wav  = tid >> 5;
  const int lane = tid & 31;
  const int hlf  = lane >> 4;
  const int l16  = lane & 15;

  const int qb = blockIdx.x & 3;
  const int h  = (blockIdx.x >> 2) & 7;
  const int b  = blockIdx.x >> 5;

  const size_t kbase = (((size_t)b * NH + h) * NTKV) << 7;  // K  [t][d] plane
  const size_t vbase = (((size_t)b * NH + h) * NHD) * NTKV; // V^T [d][t] plane

  // Issue one 64-col KV tile's async copies into buffer bufi.
  auto stage = [&](int kt, int bufi) {
    const __bf16* kg = K  + kbase + ((size_t)(kt * 64) << 7);
    const __bf16* vg = Vt + vbase + kt * 64;
    #pragma unroll
    for (int j = 0; j < 4; ++j) {                 // K tile: 64 x 128
      int c = tid * 4 + j;
      int r = c >> 4, dc = (c & 15) * 8;
      cp16_async(&kg[(size_t)r * 128 + dc], &Ks[bufi][r * 128 + dc]);
    }
    #pragma unroll
    for (int j = 0; j < 4; ++j) {                 // V^T tile: 128 x 64
      int c = tid * 4 + j;
      int d = c >> 3, t8 = (c & 7) * 8;
      cp16_async(&vg[(size_t)d * NTKV + t8], &Vs[bufi][d * 64 + t8]);
    }
  };

  const int qrow_a = qb * 128 + wav * 16 + l16;      // row for A-fragments

  // Q fragments resident for the whole KV sweep (4 k-chunks of 32).
  Frag16 qf[4];
  {
    const __bf16* qp = Q + ((((size_t)b * NH + h) * NTQ + qrow_a) << 7);
    #pragma unroll
    for (int kc = 0; kc < 4; ++kc) {
      const __bf16* p = qp + kc * 32 + hlf * 8;
      qf[kc].q[0] = *(const uint4*)p;
      qf[kc].q[1] = *(const uint4*)(p + 16);
    }
  }

  float mrow[8], lrow[8];
  v8f o[8] = {};
  unsigned vb = 0;
  #pragma unroll
  for (int i = 0; i < 8; ++i) { mrow[i] = -3.0e38f; lrow[i] = 0.f; }

  stage(0, 0);

  for (int kt = 0; kt < NTKV / 64; ++kt) {
    const int bufi = kt & 1;
    const int kv0  = kt * 64;
    wait_async_all();     // this wave's tile copies complete
    __syncthreads();      // all waves' copies complete; prev buffer free
    if (kt + 1 < NTKV / 64) stage(kt + 1, bufi ^ 1);

    // ---- S = Q * K^T  (4 n-tiles x 4 k-chunks) ----
    v8f s[4] = {};
    #pragma unroll
    for (int nt = 0; nt < 4; ++nt) {
      #pragma unroll
      for (int kc = 0; kc < 4; ++kc) {
        Frag16 bfk;
        const __bf16* p = &Ks[bufi][(nt * 16 + l16) * 128 + kc * 32 + hlf * 8];
        bfk.q[0] = *(const uint4*)p;
        bfk.q[1] = *(const uint4*)(p + 16);
        s[nt] = WMMA_BF16(qf[kc].f, bfk.f, s[nt]);
      }
    }

    // ---- mask: scores -> -1e30, track per-row validity ----
    #pragma unroll
    for (int nt = 0; nt < 4; ++nt) {
      int c = kv0 + nt * 16 + l16;
      #pragma unroll
      for (int i = 0; i < 8; ++i) {
        int qr = qb * 128 + wav * 16 + i + 8 * hlf;
        unsigned char mb = mask[((size_t)b * NTQ + qr) * NTKV + c];
        if (mb) vb |= (1u << i); else s[nt][i] = -1.0e30f;
      }
    }

    // ---- online softmax (butterfly across the 16-lane half) ----
    float alpha[8];
    #pragma unroll
    for (int i = 0; i < 8; ++i) {
      float t = fmaxf(fmaxf(s[0][i], s[1][i]), fmaxf(s[2][i], s[3][i]));
      #pragma unroll
      for (int m = 1; m < 16; m <<= 1) t = fmaxf(t, __shfl_xor(t, m, 32));
      float mnew = fmaxf(mrow[i], t);
      alpha[i] = __expf(mrow[i] - mnew);
      mrow[i] = mnew;
    }
    #pragma unroll
    for (int nt = 0; nt < 4; ++nt)
      #pragma unroll
      for (int i = 0; i < 8; ++i)
        s[nt][i] = __expf(s[nt][i] - mrow[i]);
    #pragma unroll
    for (int i = 0; i < 8; ++i) {
      float rs = s[0][i] + s[1][i] + s[2][i] + s[3][i];
      #pragma unroll
      for (int m = 1; m < 16; m <<= 1) rs += __shfl_xor(rs, m, 32);
      lrow[i] = lrow[i] * alpha[i] + rs;
    }
    #pragma unroll
    for (int t = 0; t < 8; ++t)
      #pragma unroll
      for (int i = 0; i < 8; ++i)
        o[t][i] *= alpha[i];

    // ---- stage P (accum layout -> LDS), reload as A-fragments ----
    #pragma unroll
    for (int nt = 0; nt < 4; ++nt)
      #pragma unroll
      for (int i = 0; i < 8; ++i)
        Ps[wav][(i + 8 * hlf) * 64 + nt * 16 + l16] = (__bf16)s[nt][i];
    // same-wave DS ops are in-order: no barrier needed (Ps is wave-private)

    Frag16 pf[2];
    #pragma unroll
    for (int kc = 0; kc < 2; ++kc) {
      const __bf16* p = &Ps[wav][l16 * 64 + kc * 32 + hlf * 8];
      pf[kc].q[0] = *(const uint4*)p;
      pf[kc].q[1] = *(const uint4*)(p + 16);
    }
    // ---- O += P * V ----
    #pragma unroll
    for (int t = 0; t < 8; ++t) {
      #pragma unroll
      for (int kc = 0; kc < 2; ++kc) {
        Frag16 vf;
        const __bf16* p = &Vs[bufi][(t * 16 + l16) * 64 + kc * 32 + hlf * 8];
        vf.q[0] = *(const uint4*)p;
        vf.q[1] = *(const uint4*)(p + 16);
        o[t] = WMMA_BF16(pf[kc].f, vf.f, o[t]);
      }
    }
  }

  // ---- finalize: 1/l, wipe fully-masked rows, write bf16 head-concat ----
  #pragma unroll
  for (int m = 1; m < 16; m <<= 1) vb |= (unsigned)__shfl_xor((int)vb, m, 32);
  #pragma unroll
  for (int i = 0; i < 8; ++i) {
    float inv = ((vb >> i) & 1u) ? (1.0f / lrow[i]) : 0.0f;
    int qr = qb * 128 + wav * 16 + i + 8 * hlf;
    #pragma unroll
    for (int t = 0; t < 8; ++t) {
      AO[((size_t)b * NTQ + qr) * ND + h * NHD + t * 16 + l16] =
          (__bf16)(o[t][i] * inv);
    }
  }
}

// ---------------------------------------------------------------------------
// Host-side orchestration (graph-capture safe: launches only).
// ---------------------------------------------------------------------------
extern "C" void kernel_launch(void* const* d_in, const int* in_sizes, int n_in,
                              void* d_out, int out_size, void* d_ws, size_t ws_size,
                              hipStream_t stream) {
  (void)in_sizes; (void)n_in; (void)out_size; (void)ws_size;

  const float* xq  = (const float*)d_in[0];
  const float* xkv = (const float*)d_in[1];
  const unsigned char* msk = (const unsigned char*)d_in[2];
  const float* Wq = (const float*)d_in[3];
  const float* bq = (const float*)d_in[4];
  const float* Wk = (const float*)d_in[5];
  const float* bk = (const float*)d_in[6];
  const float* Wv = (const float*)d_in[7];
  const float* bv = (const float*)d_in[8];
  const float* Wo = (const float*)d_in[9];
  const float* bo = (const float*)d_in[10];

  // workspace layout (bf16 intermediates)
  char* ws = (char*)d_ws;
  const size_t QB = (size_t)NB * NH * NTQ  * NHD;   //  2M elems
  const size_t KB = (size_t)NB * NH * NTKV * NHD;   // 16M elems
  __bf16* Qb = (__bf16*)ws;                                  ws += QB * 2;
  __bf16* Kb = (__bf16*)ws;                                  ws += KB * 2;
  __bf16* Vb = (__bf16*)ws;                                  ws += KB * 2;   // V^T
  __bf16* AO = (__bf16*)ws;

  const float qscale = 0.08838834764831845f;  // 1/sqrt(128)

  // Q/K projections (head-split bf16), V projection written transposed.
  gemm_bf16_wmma<float, 0><<<dim3(NB * NTQ  / 128, ND / 128), 256, 0, stream>>>(
      xq,  Wq, bq, (void*)Qb, NTQ,  qscale);
  gemm_bf16_wmma<float, 0><<<dim3(NB * NTKV / 128, ND / 128), 256, 0, stream>>>(
      xkv, Wk, bk, (void*)Kb, NTKV, 1.0f);
  gemm_bf16_wmma<float, 2><<<dim3(NB * NTKV / 128, ND / 128), 256, 0, stream>>>(
      xkv, Wv, bv, (void*)Vb, NTKV, 1.0f);

  // attention (B*H*(TQ/128) = 128 blocks)
  attn_fwd_wmma<<<NB * NH * (NTQ / 128), 256, 0, stream>>>(Qb, Kb, Vb, msk, AO);

  // output projection -> f32
  gemm_bf16_wmma<__bf16, 1><<<dim3(NB * NTQ / 128, ND / 128), 256, 0, stream>>>(
      AO, Wo, bo, d_out, NTQ, 1.0f);
}
